// MultiHeadAttention_16604343566715
// MI455X (gfx1250) — matrix operand order repaired and checked by reading, not compiled, not run on hardware
//
#include <hip/hip_runtime.h>
#include <hip/hip_bf16.h>
#include <math.h>
#include <cstdint>

typedef __attribute__((ext_vector_type(16))) __bf16   v16bf;
typedef __attribute__((ext_vector_type(8)))  __bf16   bf16x8;
typedef __attribute__((ext_vector_type(8)))  float    v8f;
typedef __attribute__((ext_vector_type(4)))  float    f32x4;
typedef __attribute__((ext_vector_type(4)))  uint32_t u32x4;
typedef __attribute__((ext_vector_type(8)))  uint32_t u32x8;

static constexpr int BATCH = 8, SEQ = 1024, DIM = 1024, HEADS = 16, DH = 64;
static constexpr float WIN_C = 0.17677669529663687f; // 1/(4*sqrt(2)) : 2*ndtr(-d/4)=erfc(d/(4sqrt2))

// ---- WMMA helpers ------------------------------------------------------
__device__ __forceinline__ v8f wmma_bf16(v16bf a, v16bf b, v8f c) {
    return __builtin_amdgcn_wmma_f32_16x16x32_bf16(false, a, false, b, (short)0, c, false, false);
}

// A-matrix 16x32 fragment: lane l holds row M=l%16; elements 0..7 = K half*8..+7,
// elements 8..15 = K 16+half*8..+7 (ISA 7.12.2). 'row' points at row M (row-major, bf16).
__device__ __forceinline__ v16bf make_frag_a(const __bf16* row, int half) {
    bf16x8 lo = *(const bf16x8*)(row + half * 8);
    bf16x8 hi = *(const bf16x8*)(row + 16 + half * 8);
    return __builtin_shufflevector(lo, hi, 0,1,2,3,4,5,6,7,8,9,10,11,12,13,14,15);
}

// B-matrix 32x16 fragment from [n][k] row-major tile: lane l holds col N=l%16,
// elements 0..7 = K half*8..+7, elements 8..15 = K 16+half*8..+7 (matches A).
__device__ __forceinline__ v16bf make_frag_b(const __bf16* rowN, int half) {
    bf16x8 lo = *(const bf16x8*)(rowN + half * 8);
    bf16x8 hi = *(const bf16x8*)(rowN + 16 + half * 8);
    return __builtin_shufflevector(lo, hi, 0,1,2,3,4,5,6,7,8,9,10,11,12,13,14,15);
}

// Load 16 elements (fp32 or bf16 source) -> 16 bf16 into LDS (32B, vectorized).
template <typename TX>
__device__ __forceinline__ void load16_cvt(const TX* __restrict__ src, __bf16* __restrict__ dst) {
    bf16x8 o0, o1;
    if constexpr (sizeof(TX) == 2) {
        o0 = ((const bf16x8*)src)[0];
        o1 = ((const bf16x8*)src)[1];
    } else {
        const f32x4* s4 = (const f32x4*)src;
        f32x4 a = s4[0], b = s4[1], c = s4[2], d = s4[3];
        o0[0]=(__bf16)a[0]; o0[1]=(__bf16)a[1]; o0[2]=(__bf16)a[2]; o0[3]=(__bf16)a[3];
        o0[4]=(__bf16)b[0]; o0[5]=(__bf16)b[1]; o0[6]=(__bf16)b[2]; o0[7]=(__bf16)b[3];
        o1[0]=(__bf16)c[0]; o1[1]=(__bf16)c[1]; o1[2]=(__bf16)c[2]; o1[3]=(__bf16)c[3];
        o1[4]=(__bf16)d[0]; o1[5]=(__bf16)d[1]; o1[6]=(__bf16)d[2]; o1[7]=(__bf16)d[3];
    }
    ((bf16x8*)dst)[0] = o0;
    ((bf16x8*)dst)[1] = o1;
}

// ---- CDNA5 async & TDM data movement ----------------------------------
// Per-lane async copy: global -> LDS, 16 bytes/lane, tracked by ASYNCcnt.
__device__ __forceinline__ void async_copy_b128(void* lds_ptr, const void* gptr) {
    const uint32_t lds_addr = (uint32_t)(uintptr_t)lds_ptr;  // LDS aperture: low 32 bits = offset
    asm volatile("global_load_async_to_lds_b128 %0, %1, off"
                 :: "v"(lds_addr), "v"(gptr) : "memory");
}
__device__ __forceinline__ void wait_async_le0() { asm volatile("s_wait_asynccnt 0x0" ::: "memory"); }
__device__ __forceinline__ void wait_async_le2() { asm volatile("s_wait_asynccnt 0x2" ::: "memory"); }

// Tensor Data Mover: one wave issues a 2-D tile load (bf16) described by a D#.
// Groups per ISA 8.3-8.6; data_size=1 (2 bytes), tile rows packed contiguously in LDS.
__device__ __forceinline__ void tdm_load_tile_2d(uint32_t lds_addr, const void* gaddr,
                                                 uint32_t tensor_d0, uint32_t tensor_d1,
                                                 uint32_t tile_d0, uint32_t tile_d1,
                                                 uint32_t stride_d0) {
    const uint64_t ga = (uint64_t)(uintptr_t)gaddr;
    u32x4 g0;
    g0[0] = 1u;                                              // count=1, user descriptor
    g0[1] = lds_addr;                                        // lds_addr (bytes)
    g0[2] = (uint32_t)ga;                                    // global_addr[31:0]
    g0[3] = (uint32_t)((ga >> 32) & 0x1FFFFFFu) | (2u << 30); // addr[56:32] | type=2
    u32x8 g1;
    g1[0] = (1u << 16);                                      // workgroup_mask=0, data_size=2B
    g1[1] = (tensor_d0 & 0xFFFFu) << 16;                     // tensor_dim0[15:0] @ bits63:48
    g1[2] = ((tensor_d0 >> 16) & 0xFFFFu) | ((tensor_d1 & 0xFFFFu) << 16);
    g1[3] = ((tensor_d1 >> 16) & 0xFFFFu) | ((tile_d0 & 0xFFFFu) << 16);
    g1[4] = (tile_d1 & 0xFFFFu);                             // tile_dim1; tile_dim2=0 (2D)
    g1[5] = stride_d0;                                       // tensor_dim0_stride[31:0]
    g1[6] = 0;                                               // stride hi / dim1_stride lo
    g1[7] = 0;
    u32x4 g2; g2[0] = 1u; g2[1] = 1u; g2[2] = 0u; g2[3] = 0u; // dim2/3 unused (tile_dim3=0)
    u32x4 g3; g3[0] = 0u; g3[1] = 0u; g3[2] = 0u; g3[3] = 0u;
    asm volatile("tensor_load_to_lds %0, %1, %2, %3"
                 :: "s"(g0), "s"(g1), "s"(g2), "s"(g3) : "memory");
}

// ---- Projection GEMM: Y = (X * W^T + bias) * scale ---------------------
// Block = 128 threads (4 waves); block tile 128x64, K-step 64; 16 WMMA/wave/step.
// MODE 0: store bf16 at [b,h,s,dh]      (K and Q)
// MODE 1: store bf16 at [b,h,dh,s]      (V transposed for attn*V B-frags)
// MODE 2: store fp32  at [m, e]         (final output)
template <typename TX, int MODE>
__global__ __launch_bounds__(128) void proj_kernel(
    const TX* __restrict__ X, const float* __restrict__ W,
    const float* __restrict__ bias, void* __restrict__ Yv, float scale)
{
    __shared__ alignas(16) __bf16 As[128 * 64];  // [m][k]
    __shared__ alignas(16) __bf16 Bs[64 * 64];   // [n][k]  (rows of W directly)

    const int t = threadIdx.x;
    const int lane = t & 31, wave = t >> 5;
    const int half = lane >> 4, lm = lane & 15;
    const int gm = blockIdx.x * 128;
    const int gn = blockIdx.y * 64;

    v8f acc[2][4] = {};

    for (int k0 = 0; k0 < DIM; k0 += 64) {
        // A tile: thread t loads one full 64-elem row
        const TX* xr = X + (size_t)(gm + t) * DIM + k0;
#pragma unroll
        for (int c = 0; c < 4; ++c) load16_cvt<TX>(xr + c * 16, As + t * 64 + c * 16);
        // B tile: 64 rows of W, 32 elems per thread
        const float* wr = W + (size_t)(gn + (t >> 1)) * DIM + k0 + (t & 1) * 32;
        load16_cvt<float>(wr,      Bs + (t >> 1) * 64 + (t & 1) * 32);
        load16_cvt<float>(wr + 16, Bs + (t >> 1) * 64 + (t & 1) * 32 + 16);
        __syncthreads();

#pragma unroll
        for (int kc = 0; kc < 2; ++kc) {
            const v16bf a0 = make_frag_a(As + (wave * 32 + lm) * 64 + kc * 32, half);
            const v16bf a1 = make_frag_a(As + (wave * 32 + 16 + lm) * 64 + kc * 32, half);
#pragma unroll
            for (int ns = 0; ns < 4; ++ns) {
                v16bf b = make_frag_b(Bs + (ns * 16 + lm) * 64 + kc * 32, half);
                acc[0][ns] = wmma_bf16(a0, b, acc[0][ns]);
                acc[1][ns] = wmma_bf16(a1, b, acc[1][ns]);
            }
        }
        __syncthreads();
    }

#pragma unroll
    for (int ns = 0; ns < 4; ++ns) {
        const int n = gn + ns * 16 + lm;
        const float bv = bias[n];
#pragma unroll
        for (int mi = 0; mi < 2; ++mi) {
#pragma unroll
            for (int r = 0; r < 8; ++r) {
                const int m = gm + wave * 32 + mi * 16 + r + half * 8;
                const float y = (acc[mi][ns][r] + bv) * scale;
                if constexpr (MODE == 2) {
                    ((float*)Yv)[(size_t)m * DIM + n] = y;
                } else {
                    const int b = m >> 10, s = m & 1023;   // m = b*SEQ + s
                    const int h = n >> 6,  d = n & 63;     // n = h*DH + d
                    size_t idx;
                    if constexpr (MODE == 0)
                        idx = (((size_t)(b * HEADS + h)) * SEQ + s) * DH + d;
                    else
                        idx = (((size_t)(b * HEADS + h)) * DH + d) * SEQ + s;
                    ((__bf16*)Yv)[idx] = (__bf16)y;
                }
            }
        }
    }
}

// ---- Windowed flash attention -----------------------------------------
// grid = (SEQ/64, BATCH*HEADS), block = 128 (4 waves, 16 query rows/wave).
// K tiles staged by the Tensor Data Mover (TENSORcnt), V tiles by per-lane
// async copies (ASYNCcnt); both double-buffered so DMA overlaps WMMA/softmax.
__global__ __launch_bounds__(128) void attn_kernel(
    const __bf16* __restrict__ Q,   // [b,h,s,dh], pre-scaled by 1/8
    const __bf16* __restrict__ K,   // [b,h,s,dh]
    const __bf16* __restrict__ Vt,  // [b,h,dh,s]
    const unsigned char* __restrict__ mask,  // [b, s]
    __bf16* __restrict__ ctx)       // [b, s, D]  (D col = h*64+d)
{
    __shared__ alignas(16) __bf16 Ks[2][32 * 64];   // [key][d], double buffered
    __shared__ alignas(16) __bf16 Vs[2][64 * 32];   // [d][key], double buffered
    __shared__ alignas(16) __bf16 Ps[4][16 * 32];   // per-wave prob tile [qrow][key]
    __shared__ float wtab[SEQ];                     // wtab[d] = erfc(d/(4*sqrt2))

    const int t = threadIdx.x;
    const int lane = t & 31, wave = t >> 5;
    const int half = lane >> 4, lm = lane & 15;
    const int bh = blockIdx.y;
    const int b = bh >> 4;                  // / HEADS
    const int h = bh & 15;
    const int q0 = blockIdx.x * 64 + wave * 16;

    // Distance-only Gaussian window table (visible after first in-loop barrier).
    for (int i = t; i < SEQ; i += 128) wtab[i] = erfcf((float)i * WIN_C);

    const __bf16* qbase = Q + ((size_t)bh * SEQ + q0) * DH;
    const __bf16* kbase = K + (size_t)bh * SEQ * DH;
    const __bf16* vbase = Vt + (size_t)bh * DH * SEQ;

    // Stage key/value tile for keys [j0, j0+32) into buffer `bi`.
    auto stage_kv = [&](int bi, int j0) {
        if (t < 32) {  // one wave drives the TDM for the K tile (EXEC ignored by TDM)
            tdm_load_tile_2d((uint32_t)(uintptr_t)&Ks[bi][0], kbase + (size_t)j0 * DH,
                             /*tensor_d0=*/DH, /*tensor_d1=*/SEQ,
                             /*tile_d0=*/DH, /*tile_d1=*/32, /*stride_d0=*/DH);
        }
        const int vr = t >> 1, vc = (t & 1) * 16;
        __bf16* vd = &Vs[bi][vr * 32 + vc];
        const __bf16* vsrc = vbase + (size_t)vr * SEQ + j0 + vc;
        async_copy_b128(vd, vsrc);
        async_copy_b128(vd + 8, vsrc + 8);
    };

    // Q A-fragments (rows lm of this wave's 16-row block), K-dim = dh = 64
    const v16bf qa0 = make_frag_a(qbase + lm * DH + 0, half);
    const v16bf qa1 = make_frag_a(qbase + lm * DH + 32, half);

    float mrow[8], lrow[8];
#pragma unroll
    for (int r = 0; r < 8; ++r) { mrow[r] = -3.0e38f; lrow[r] = 0.0f; }
    v8f acc[4] = {};

    stage_kv(0, 0);
    int buf = 0;

    for (int j0 = 0; j0 < SEQ; j0 += 32) {
        const bool has_next = (j0 + 32) < SEQ;
        if (has_next) stage_kv(buf ^ 1, j0 + 32);
        if (t < 32) {
            if (has_next) __builtin_amdgcn_s_wait_tensorcnt(1);
            else          __builtin_amdgcn_s_wait_tensorcnt(0);
        }
        if (has_next) wait_async_le2(); else wait_async_le0();
        __syncthreads();

        const __bf16* ks = &Ks[buf][0];
        const __bf16* vs = &Vs[buf][0];

        // scores: two 16x16 tiles over keys j0..j0+31, K-dim 64 (2 chained WMMAs each)
        v8f c0 = {}, c1 = {};
        c0 = wmma_bf16(qa0, make_frag_b(ks + lm * 64 + 0, half), c0);
        c0 = wmma_bf16(qa1, make_frag_b(ks + lm * 64 + 32, half), c0);
        c1 = wmma_bf16(qa0, make_frag_b(ks + (16 + lm) * 64 + 0, half), c1);
        c1 = wmma_bf16(qa1, make_frag_b(ks + (16 + lm) * 64 + 32, half), c1);

        const int k0g = j0 + lm, k1g = j0 + 16 + lm;
        const float msk0 = mask[b * SEQ + k0g] ? 1.0f : 0.0f;
        const float msk1 = mask[b * SEQ + k1g] ? 1.0f : 0.0f;

#pragma unroll
        for (int r = 0; r < 8; ++r) {
            const int qrow = q0 + r + half * 8;
            const float w0 = wtab[__builtin_abs(qrow - k0g)];
            const float w1 = wtab[__builtin_abs(qrow - k1g)];
            const float s0 = (msk0 != 0.0f ? -1.0e18f : c0[r]) * w0;
            const float s1 = (msk1 != 0.0f ? -1.0e18f : c1[r]) * w1;

            // row max across the 16 lanes of each half (row m = r + 8*half)
            float mx = fmaxf(s0, s1);
#pragma unroll
            for (int off = 1; off < 16; off <<= 1) mx = fmaxf(mx, __shfl_xor(mx, off, 32));

            const float mnew = fmaxf(mrow[r], mx);
            const float sc = __expf(mrow[r] - mnew);
            mrow[r] = mnew;
            float e0 = __expf(s0 - mnew);
            float e1 = __expf(s1 - mnew);
            float sum = e0 + e1;
#pragma unroll
            for (int off = 1; off < 16; off <<= 1) sum += __shfl_xor(sum, off, 32);
            lrow[r] = lrow[r] * sc + sum;
#pragma unroll
            for (int ns = 0; ns < 4; ++ns) acc[ns][r] *= sc;

            // masked probabilities -> LDS (C layout -> A layout bounce)
            e0 *= (1.0f - msk0);
            e1 *= (1.0f - msk1);
            Ps[wave][(r + half * 8) * 32 + lm] = (__bf16)e0;
            Ps[wave][(r + half * 8) * 32 + 16 + lm] = (__bf16)e1;
        }

        // ctx += P(16x32) * V(32x64)
        const v16bf pa = make_frag_a(Ps[wave] + lm * 32, half);
#pragma unroll
        for (int ns = 0; ns < 4; ++ns) {
            v16bf vb = make_frag_b(vs + (ns * 16 + lm) * 32, half);
            acc[ns] = wmma_bf16(pa, vb, acc[ns]);
        }
        __syncthreads();
        buf ^= 1;
    }

    // normalize and write ctx (bf16, [b][s][h*64+d])
#pragma unroll
    for (int r = 0; r < 8; ++r) {
        const float inv = 1.0f / lrow[r];
        const int srow = q0 + r + half * 8;
#pragma unroll
        for (int ns = 0; ns < 4; ++ns) {
            const int dcol = ns * 16 + lm;
            ctx[((size_t)b * SEQ + srow) * DIM + h * DH + dcol] = (__bf16)(acc[ns][r] * inv);
        }
    }
}

// ---- Host launch -------------------------------------------------------
extern "C" void kernel_launch(void* const* d_in, const int* in_sizes, int n_in,
                              void* d_out, int out_size, void* d_ws, size_t ws_size,
                              hipStream_t stream)
{
    const float* key   = (const float*)d_in[0];
    const float* value = (const float*)d_in[1];
    const float* query = (const float*)d_in[2];
    const float* Wk    = (const float*)d_in[3];
    const float* bk    = (const float*)d_in[4];
    const float* Wv    = (const float*)d_in[5];
    const float* bv    = (const float*)d_in[6];
    const float* Wq    = (const float*)d_in[7];
    const float* bq    = (const float*)d_in[8];
    const float* Wo    = (const float*)d_in[9];
    const float* bo    = (const float*)d_in[10];
    const unsigned char* mask = (const unsigned char*)d_in[11];

    const size_t perTensor = (size_t)BATCH * HEADS * SEQ * DH;  // 8M bf16 elems
    __bf16* q_ws = (__bf16*)d_ws;
    __bf16* k_ws = q_ws + perTensor;
    __bf16* v_ws = k_ws + perTensor;
    __bf16* c_ws = v_ws + perTensor;

    dim3 pgrid(BATCH * SEQ / 128, DIM / 64);  // (64, 16)
    dim3 blk(128);

    proj_kernel<float, 0><<<pgrid, blk, 0, stream>>>(key,   Wk, bk, (void*)k_ws, 1.0f);
    proj_kernel<float, 1><<<pgrid, blk, 0, stream>>>(value, Wv, bv, (void*)v_ws, 1.0f);
    proj_kernel<float, 0><<<pgrid, blk, 0, stream>>>(query, Wq, bq, (void*)q_ws, 0.125f); // 1/sqrt(64)

    dim3 agrid(SEQ / 64, BATCH * HEADS);     // (16, 128)
    attn_kernel<<<agrid, blk, 0, stream>>>(q_ws, k_ws, v_ws, mask, c_ws);

    proj_kernel<__bf16, 2><<<pgrid, blk, 0, stream>>>(c_ws, Wo, bo, d_out, 1.0f);
}
